// ExpertRouter_46007689675044
// MI455X (gfx1250) — matrix operand order repaired and checked by reading, not compiled, not run on hardware
//
#include <hip/hip_runtime.h>
#include <hip/hip_bf16.h>

// ---------------------------------------------------------------------------
// MoE top-2 FFN for MI455X (gfx1250, wave32, WMMA bf16 16x16x32)
// M=32 token tiles: each B fragment feeds two WMMA accumulator chains.
// ---------------------------------------------------------------------------

typedef __bf16 bf16_t;
typedef __attribute__((ext_vector_type(16))) __bf16 v16bf;
typedef __attribute__((ext_vector_type(8)))  __bf16 v8bf;
typedef __attribute__((ext_vector_type(8)))  float  v8f;

#define D_MODEL 1024
#define D_FF    2048
#define N_EXP   8
#define N_TOK   8192          // B*S = 4*2048
#define CAP     8192          // max tokens per expert (each token at most once)
#define MTILE   32            // tokens per workgroup in the FFN kernel

// ---------------------------------------------------------------------------
// Kernel 0: zero output + expert counters
// ---------------------------------------------------------------------------
__global__ __launch_bounds__(256)
void zero_kernel(float* __restrict__ out, int n, int* __restrict__ counts) {
    int i = blockIdx.x * blockDim.x + threadIdx.x;
    if (i < n) out[i] = 0.0f;
    if (i < N_EXP) counts[i] = 0;
}

// ---------------------------------------------------------------------------
// Kernel 1: gating + top-2 softmax + scatter into per-expert lists.
// One wave (32 lanes) per token; also converts x row to bf16.
// ---------------------------------------------------------------------------
__global__ __launch_bounds__(256)
void gate_scatter_kernel(const float* __restrict__ x,
                         const float* __restrict__ gw,   // [D_MODEL][N_EXP]
                         const float* __restrict__ gb,   // [N_EXP]
                         bf16_t*      __restrict__ x16,  // [N_TOK][D_MODEL]
                         int*         __restrict__ perm_tok,  // [N_EXP][CAP]
                         float*       __restrict__ perm_wgt,  // [N_EXP][CAP]
                         int*         __restrict__ counts)    // [N_EXP]
{
    int wave = threadIdx.x >> 5;
    int lane = threadIdx.x & 31;
    int tok  = blockIdx.x * 8 + wave;
    if (tok >= N_TOK) return;

    const float* xr = x + (size_t)tok * D_MODEL;
    float acc[N_EXP];
#pragma unroll
    for (int e = 0; e < N_EXP; ++e) acc[e] = 0.0f;

    for (int d = lane; d < D_MODEL; d += 32) {
        float xv = xr[d];
        x16[(size_t)tok * D_MODEL + d] = (bf16_t)xv;     // fused bf16 convert
        const float* g = gw + (size_t)d * N_EXP;
#pragma unroll
        for (int e = 0; e < N_EXP; ++e) acc[e] += xv * g[e];
    }
    // wave32 reduction
#pragma unroll
    for (int e = 0; e < N_EXP; ++e) {
#pragma unroll
        for (int s = 16; s >= 1; s >>= 1) acc[e] += __shfl_xor(acc[e], s, 32);
    }

    if (lane == 0) {
        float v[N_EXP];
#pragma unroll
        for (int e = 0; e < N_EXP; ++e) v[e] = acc[e] + gb[e];
        // top-2, strict '>' keeps lowest index on ties (matches jax top_k)
        int i0 = 0;
#pragma unroll
        for (int e = 1; e < N_EXP; ++e) if (v[e] > v[i0]) i0 = e;
        int i1 = (i0 == 0) ? 1 : 0;
#pragma unroll
        for (int e = 0; e < N_EXP; ++e)
            if (e != i0 && v[e] > v[i1]) i1 = e;
        // softmax over the two kept logits (others underflow to exactly 0)
        float m  = v[i0];
        float e0 = __expf(v[i0] - m);
        float e1 = __expf(v[i1] - m);
        float inv = 1.0f / (e0 + e1);
        float w0 = e0 * inv, w1 = e1 * inv;

        int p0 = atomicAdd(&counts[i0], 1);
        perm_tok[i0 * CAP + p0] = tok;
        perm_wgt[i0 * CAP + p0] = w0;
        int p1 = atomicAdd(&counts[i1], 1);
        perm_tok[i1 * CAP + p1] = tok;
        perm_wgt[i1 * CAP + p1] = w1;
    }
}

// ---------------------------------------------------------------------------
// Kernel 2: per-expert transpose + fp32->bf16 convert.
// src: [E][R][C] fp32 row-major  ->  dst: [E][C][R] bf16 row-major
// ---------------------------------------------------------------------------
__global__ __launch_bounds__(256)
void transpose_cvt_kernel(const float* __restrict__ src,
                          bf16_t*      __restrict__ dst,
                          int R, int C)
{
    __shared__ float tile[32][33];
    const float* s = src + (size_t)blockIdx.z * R * C;
    bf16_t*      d = dst + (size_t)blockIdx.z * R * C;
    int c0 = blockIdx.x * 32, r0 = blockIdx.y * 32;
    int tx = threadIdx.x & 31, ty = threadIdx.x >> 5;   // 32 x 8
#pragma unroll
    for (int i = 0; i < 32; i += 8)
        tile[ty + i][tx] = s[(size_t)(r0 + ty + i) * C + c0 + tx];
    __syncthreads();
#pragma unroll
    for (int i = 0; i < 32; i += 8)
        d[(size_t)(c0 + ty + i) * R + r0 + tx] = (bf16_t)tile[tx][ty + i];
}

// ---------------------------------------------------------------------------
// Kernel 3: grouped expert FFN with WMMA bf16, M=32 tiles.
// grid = (CAP/32, N_EXP); block = 256 (8 waves). Each block: 32 tokens of
// one expert. Phase A: h = relu(x @ w1 + b1) -> LDS (bf16).
// Phase B: y = h @ w2 + b2, scaled by routing weight, atomically combined.
// Dynamic LDS: xs 64KB + hs 128KB + token/weight arrays (WGP has 320KB).
// ---------------------------------------------------------------------------
__global__ __launch_bounds__(256)
void moe_ffn_kernel(const bf16_t* __restrict__ x16,   // [N_TOK][D_MODEL]
                    const bf16_t* __restrict__ w1t,   // [E][D_FF][D_MODEL]
                    const bf16_t* __restrict__ w2t,   // [E][D_MODEL][D_FF]
                    const float*  __restrict__ b1,    // [E][D_FF]
                    const float*  __restrict__ b2,    // [E][D_MODEL]
                    const int*    __restrict__ perm_tok,
                    const float*  __restrict__ perm_wgt,
                    const int*    __restrict__ counts,
                    float*        __restrict__ out)   // [N_TOK][D_MODEL]
{
    int e    = blockIdx.y;
    int tile = blockIdx.x;
    int cnt  = counts[e];
    if (tile * MTILE >= cnt) return;

    extern __shared__ char smem[];
    bf16_t* xs    = (bf16_t*)smem;                        // 32*1024*2 =  64 KB
    bf16_t* hs    = (bf16_t*)(smem + 65536);              // 32*2048*2 = 128 KB
    int*    s_tok = (int*)  (smem + 65536 + 131072);
    float*  s_wgt = (float*)(smem + 65536 + 131072 + 128);

    int tid  = threadIdx.x;
    int lane = tid & 31;
    int wave = tid >> 5;

    if (tid < MTILE) {
        int p = tile * MTILE + tid;
        int t = (p < cnt) ? perm_tok[e * CAP + p] : -1;
        s_tok[tid] = t;
        s_wgt[tid] = (p < cnt) ? perm_wgt[e * CAP + p] : 0.0f;
    }
    __syncthreads();

    // --- gather 32 token rows of x (bf16) into LDS ---
    {
        int row = tid >> 3;           // 0..31
        int col = (tid & 7) * 128;    // 128 halves per thread (256 B)
        int t = s_tok[row]; if (t < 0) t = 0;   // harmless dummy row
        const uint4* srcp = (const uint4*)(x16 + (size_t)t * D_MODEL + col);
        uint4*       dstp = (uint4*)(xs + row * D_MODEL + col);
#pragma unroll
        for (int i = 0; i < 16; ++i) dstp[i] = srcp[i];
    }
    __syncthreads();

    const int ln15  = lane & 15;             // A row m / B col n
    const int koffA = (lane >> 4) * 8;       // A-fragment K offset (ISA layout)
    const int koffB = (lane >> 4) * 16;      // unused; B fragments also use koffA
    const int mbase = (lane >> 4) * 8;       // C/D row base for this lane half

    // ---------------- Phase A: h = relu(x @ w1 + b1) ----------------
    const bf16_t* w1e = w1t + (size_t)e * D_FF * D_MODEL;
    for (int j = 0; j < 16; ++j) {                 // 128 ftiles over 8 waves
        int fcol = (wave + 8 * j) * 16;
        v8f c0 = {}, c1 = {};
        for (int k0 = 0; k0 < D_MODEL; k0 += 32) {
            const bf16_t* bp = w1e + (size_t)(fcol + ln15) * D_MODEL + k0 + koffA;
            v8bf bl = *(const v8bf*)bp;
            v8bf bh = *(const v8bf*)(bp + 16);
            v16bf b = __builtin_shufflevector(bl, bh,
                         0,1,2,3,4,5,6,7,8,9,10,11,12,13,14,15);

            v8bf a0l = *(const v8bf*)(xs + ln15 * D_MODEL + k0 + koffA);
            v8bf a0h = *(const v8bf*)(xs + ln15 * D_MODEL + k0 + 16 + koffA);
            v16bf a0 = __builtin_shufflevector(a0l, a0h,
                         0,1,2,3,4,5,6,7,8,9,10,11,12,13,14,15);
            v8bf a1l = *(const v8bf*)(xs + (16 + ln15) * D_MODEL + k0 + koffA);
            v8bf a1h = *(const v8bf*)(xs + (16 + ln15) * D_MODEL + k0 + 16 + koffA);
            v16bf a1 = __builtin_shufflevector(a1l, a1h,
                         0,1,2,3,4,5,6,7,8,9,10,11,12,13,14,15);

            c0 = __builtin_amdgcn_wmma_f32_16x16x32_bf16(
                     false, a0, false, b, (short)0, c0, false, false);
            c1 = __builtin_amdgcn_wmma_f32_16x16x32_bf16(
                     false, a1, false, b, (short)0, c1, false, false);
        }
        float bias = b1[(size_t)e * D_FF + fcol + ln15];
#pragma unroll
        for (int r = 0; r < 8; ++r) {
            hs[(mbase + r) * D_FF + fcol + ln15]      = (bf16_t)fmaxf(c0[r] + bias, 0.0f);
            hs[(16 + mbase + r) * D_FF + fcol + ln15] = (bf16_t)fmaxf(c1[r] + bias, 0.0f);
        }
    }
    __syncthreads();

    // ---------------- Phase B: y = h @ w2 + b2; weighted combine ----------
    const bf16_t* w2e = w2t + (size_t)e * D_MODEL * D_FF;
    for (int j = 0; j < 8; ++j) {                  // 64 ntiles over 8 waves
        int ncol = (wave + 8 * j) * 16;
        v8f c0 = {}, c1 = {};
        for (int k0 = 0; k0 < D_FF; k0 += 32) {
            const bf16_t* bp = w2e + (size_t)(ncol + ln15) * D_FF + k0 + koffA;
            v8bf bl = *(const v8bf*)bp;
            v8bf bh = *(const v8bf*)(bp + 16);
            v16bf b = __builtin_shufflevector(bl, bh,
                         0,1,2,3,4,5,6,7,8,9,10,11,12,13,14,15);

            v8bf a0l = *(const v8bf*)(hs + ln15 * D_FF + k0 + koffA);
            v8bf a0h = *(const v8bf*)(hs + ln15 * D_FF + k0 + 16 + koffA);
            v16bf a0 = __builtin_shufflevector(a0l, a0h,
                         0,1,2,3,4,5,6,7,8,9,10,11,12,13,14,15);
            v8bf a1l = *(const v8bf*)(hs + (16 + ln15) * D_FF + k0 + koffA);
            v8bf a1h = *(const v8bf*)(hs + (16 + ln15) * D_FF + k0 + 16 + koffA);
            v16bf a1 = __builtin_shufflevector(a1l, a1h,
                         0,1,2,3,4,5,6,7,8,9,10,11,12,13,14,15);

            c0 = __builtin_amdgcn_wmma_f32_16x16x32_bf16(
                     false, a0, false, b, (short)0, c0, false, false);
            c1 = __builtin_amdgcn_wmma_f32_16x16x32_bf16(
                     false, a1, false, b, (short)0, c1, false, false);
        }
        float bias = b2[(size_t)e * D_MODEL + ncol + ln15];
#pragma unroll
        for (int r = 0; r < 8; ++r) {
            int m0 = mbase + r;
            int t0 = s_tok[m0];
            if (t0 >= 0)
                unsafeAtomicAdd(out + (size_t)t0 * D_MODEL + ncol + ln15,
                                s_wgt[m0] * (c0[r] + bias));
            int m1 = 16 + mbase + r;
            int t1 = s_tok[m1];
            if (t1 >= 0)
                unsafeAtomicAdd(out + (size_t)t1 * D_MODEL + ncol + ln15,
                                s_wgt[m1] * (c1[r] + bias));
        }
    }
}

// ---------------------------------------------------------------------------
// Host-side launcher
// ---------------------------------------------------------------------------
extern "C" void kernel_launch(void* const* d_in, const int* in_sizes, int n_in,
                              void* d_out, int out_size, void* d_ws, size_t ws_size,
                              hipStream_t stream) {
    const float* x      = (const float*)d_in[0];
    const float* gate_w = (const float*)d_in[1];
    const float* gate_b = (const float*)d_in[2];
    const float* w1     = (const float*)d_in[3];
    const float* b1     = (const float*)d_in[4];
    const float* w2     = (const float*)d_in[5];
    const float* b2     = (const float*)d_in[6];
    float* out = (float*)d_out;

    // workspace carve-out (all offsets 256B-aligned)
    char* ws = (char*)d_ws;
    size_t off = 0;
    bf16_t* x16 = (bf16_t*)(ws + off); off += (size_t)N_TOK * D_MODEL * 2;        // 16 MB
    bf16_t* w1t = (bf16_t*)(ws + off); off += (size_t)N_EXP * D_FF * D_MODEL * 2; // 32 MB
    bf16_t* w2t = (bf16_t*)(ws + off); off += (size_t)N_EXP * D_MODEL * D_FF * 2; // 32 MB
    int*    perm_tok = (int*)  (ws + off); off += (size_t)N_EXP * CAP * 4;
    float*  perm_wgt = (float*)(ws + off); off += (size_t)N_EXP * CAP * 4;
    int*    counts   = (int*)  (ws + off); off += 256;
    (void)ws_size; (void)in_sizes; (void)n_in; (void)out_size;

    // 1) zero output + counters
    {
        int n = N_TOK * D_MODEL;
        zero_kernel<<<(n + 255) / 256, 256, 0, stream>>>(out, n, counts);
    }
    // 2) gating + scatter + x bf16 convert (1 wave per token)
    gate_scatter_kernel<<<N_TOK / 8, 256, 0, stream>>>(
        x, gate_w, gate_b, x16, perm_tok, perm_wgt, counts);
    // 3) w1 [E][D][F] -> w1t [E][F][D] bf16
    transpose_cvt_kernel<<<dim3(D_FF / 32, D_MODEL / 32, N_EXP), 256, 0, stream>>>(
        w1, w1t, D_MODEL, D_FF);
    // 4) w2 [E][F][D] -> w2t [E][D][F] bf16
    transpose_cvt_kernel<<<dim3(D_MODEL / 32, D_FF / 32, N_EXP), 256, 0, stream>>>(
        w2, w2t, D_FF, D_MODEL);
    // 5) grouped WMMA FFN (M=32 tiles, B-fragment reuse)
    {
        size_t lds_bytes = 65536 + 131072 + 128 + 128;   // xs + hs + tok + wgt
        dim3 grid(CAP / MTILE, N_EXP, 1);
        moe_ffn_kernel<<<grid, 256, lds_bytes, stream>>>(
            x16, w1t, w2t, b1, b2, perm_tok, perm_wgt, counts, out);
    }
}
